// GraphTCN_14113262535047
// MI455X (gfx1250) — matrix operand order repaired and checked by reading, not compiled, not run on hardware
//
#include <hip/hip_runtime.h>
#include <hip/hip_bf16.h>
#include <math.h>

// ---------------------------------------------------------------------------
// GraphTCN on MI455X (gfx1250): every MLP layer runs on v_wmma_f32_16x16x32_bf16
// (wave32, 16 rows per wave, bf16 inputs, fp32 accumulate). MLP shapes are
// template-compile-time so each kernel is a straight-line WMMA sequence (no
// runtime branches around v_wmma). Edge aggregation uses global_atomic_add_f32.
// ---------------------------------------------------------------------------

typedef __attribute__((ext_vector_type(16))) __bf16 v16bf;
typedef __attribute__((ext_vector_type(8)))  __bf16 v8bf;
typedef __attribute__((ext_vector_type(8)))  float  v8f;

#define ACT_K  64   // padded K width of the LDS activation tile (max K=40 -> 64)
#define WAVES  8    // waves per 256-thread block
#define LDSACT (WAVES * 16 * ACT_K)

struct MlpOff {                 // runtime part of an MLP: arena offsets only
    int woff[4], boff[4];
};

// ---------------- WMMA fragment helpers (wave32; A and B share contiguous k = 16h+i) ------

// A-matrix 16x32 bf16: lane holds row M = lane&15;
// element i -> K = ((lane>>4)<<4) + i. With row-major LDS act
// [16][ACT_K] this is two contiguous 16B ds_load_b128s.
__device__ __forceinline__ v16bf load_a_frag(const __bf16* act, int lane, int kbase) {
    int row   = lane & 15;
    int khalf = (lane >> 4) << 4;
    const __bf16* p = act + row * ACT_K + kbase + khalf;
    v8bf lo = *(const v8bf*)(p);        // K = kbase+khalf+0..7
    v8bf hi = *(const v8bf*)(p + 8);    // K = kbase+khalf+8..15
    v16bf a;
#pragma unroll
    for (int i = 0; i < 8; ++i) { a[i] = lo[i]; a[i + 8] = hi[i]; }
    return a;
}

// B-matrix 32x16 bf16: lane holds column N = lane&15; lanes 0-15 K=0..15,
// lanes 16-31 K=16..31. Weights packed N-major [n][Kp] -> one 32B load.
__device__ __forceinline__ v16bf load_b_frag(const __bf16* w, int Kp, int lane,
                                             int nt, int kbase) {
    int n  = (lane & 15) + nt * 16;
    int k0 = kbase + ((lane >> 4) << 4);
    return *(const v16bf*)(w + (long)n * Kp + k0);
}

__device__ __forceinline__ v8f wmma_bf16(v16bf a, v16bf b, v8f c) {
    return __builtin_amdgcn_wmma_f32_16x16x32_bf16(false, a, false, b,
                                                   (short)0, c, false, false);
}

// ---------------- generic fused-MLP executor (per-wave tile of 16 rows) ----
// Kin/Nout arrive as compile-time constants after inlining -> branches fold.

__device__ __forceinline__ void mlp_hidden_layer(const __bf16* wa, const float* ba,
                                                 int woff, int boff, int Kin, int Nout,
                                                 __bf16* act, int lane) {
    const int  Kp  = (Kin + 31) & ~31;
    const int  NT  = (Nout + 15) >> 4;          // 40 -> 3 tiles
    const bool two = (Kin > 32);
    v16bf a0 = load_a_frag(act, lane, 0);
    v16bf a1 = two ? load_a_frag(act, lane, 32) : a0;
    v8f acc[3];
#pragma unroll
    for (int nt = 0; nt < 3; ++nt) {
        if (nt < NT) {
            v8f c = {};
            v16bf b0 = load_b_frag(wa + woff, Kp, lane, nt, 0);
            c = wmma_bf16(a0, b0, c);
            if (two) {
                v16bf b1 = load_b_frag(wa + woff, Kp, lane, nt, 32);
                c = wmma_bf16(a1, b1, c);
            }
            acc[nt] = c;
        }
    }
    // bias + ReLU, write back as bf16 (C/D layout: lane = col, reg = row)
    int n = lane & 15;
    int rbase = (lane >> 4) << 3;
#pragma unroll
    for (int nt = 0; nt < 3; ++nt) {
        if (nt < NT) {
            int col = nt * 16 + n;
            float bias = ba[boff + col];      // padded cols: zero W, zero bias
#pragma unroll
            for (int r = 0; r < 8; ++r) {
                float v = acc[nt][r] + bias;
                v = v > 0.f ? v : 0.f;
                act[(rbase + r) * ACT_K + col] = (__bf16)v;
            }
        }
    }
    if (NT == 3) {                        // zero pad cols 48..63 for next K=40 layer
#pragma unroll
        for (int r = 0; r < 8; ++r)
            act[(rbase + r) * ACT_K + 48 + n] = (__bf16)0.f;
    }
}

__device__ __forceinline__ v8f mlp_final_layer(const __bf16* wa, const float* ba,
                                               int woff, int boff, int Kin,
                                               const __bf16* act, int lane) {
    const int Kp = (Kin + 31) & ~31;
    v16bf a0 = load_a_frag(act, lane, 0);
    v8f c = {};
    v16bf b0 = load_b_frag(wa + woff, Kp, lane, 0, 0);
    c = wmma_bf16(a0, b0, c);
    if (Kin > 32) {
        v16bf a1 = load_a_frag(act, lane, 32);
        v16bf b1 = load_b_frag(wa + woff, Kp, lane, 0, 32);
        c = wmma_bf16(a1, b1, c);
    }
    float bias = ba[boff + (lane & 15)];
#pragma unroll
    for (int r = 0; r < 8; ++r) c[r] += bias;
    return c;
}

// Dims chain as template pack, e.g. mlp_run<14,40,40,40,4>
template <int... Ds>
__device__ __forceinline__ v8f mlp_run(const MlpOff m, const __bf16* wa, const float* ba,
                                       __bf16* act, int lane) {
    constexpr int nd = sizeof...(Ds);
    constexpr int dims[nd] = {Ds...};
    constexpr int nL = nd - 1;
    __syncthreads();                                   // staging visible
#pragma unroll
    for (int l = 0; l < nL - 1; ++l) {
        mlp_hidden_layer(wa, ba, m.woff[l], m.boff[l], dims[l], dims[l + 1], act, lane);
        __syncthreads();
    }
    return mlp_final_layer(wa, ba, m.woff[nL - 1], m.boff[nL - 1], dims[nL - 1], act, lane);
}

// ---------------- weight repack: fp32 [K,N] -> bf16 [Npad][Kpad] -----------

struct PackLin { const float* W; const float* B; int K, N, woff, boff; };
struct PackArgs { PackLin l[72]; int n; };

__global__ __launch_bounds__(256)
void pack_kernel(PackArgs pa, __bf16* wA, float* bA) {
    int li = blockIdx.x;
    if (li >= pa.n) return;
    PackLin L = pa.l[li];
    int Kp = (L.K + 31) & ~31, Np = (L.N + 15) & ~15;
    int tot = Kp * Np;
    for (int idx = threadIdx.x; idx < tot; idx += blockDim.x) {
        int n = idx / Kp, k = idx - n * Kp;
        float v = (k < L.K && n < L.N) ? L.W[k * L.N + n] : 0.f;
        wA[L.woff + idx] = (__bf16)v;
    }
    for (int n = threadIdx.x; n < Np; n += blockDim.x)
        bA[L.boff + n] = (n < L.N) ? L.B[n] : 0.f;
}

// ---------------- kernels ---------------------------------------------------

// Dense row-wise MLP (encoders): out = relu(mlp(in))
template <int OUTSTRIDE, int OUTDIM, int... Ds>
__global__ __launch_bounds__(256)
void enc_kernel(MlpOff mlp, const __bf16* wa, const float* ba,
                const float* __restrict__ in, int nRows,
                float* __restrict__ out) {
    constexpr int dims0[] = {Ds...};
    constexpr int INDIM = dims0[0];
    __shared__ alignas(32) __bf16 actS[LDSACT];
    int lane = threadIdx.x & 31, wave = threadIdx.x >> 5;
    __bf16* act = actS + wave * 16 * ACT_K;
    long base = ((long)blockIdx.x * WAVES + wave) * 16;
    int row = lane & 15;
    long rI = base + row; if (rI >= nRows) rI = nRows - 1;
    int j0 = (lane >> 4) << 4;
#pragma unroll
    for (int j = 0; j < 16; ++j) {
        int k = j0 + j;
        float v = (k < INDIM) ? in[rI * INDIM + k] : 0.f;
        act[row * ACT_K + k] = (__bf16)v;
    }
    v8f c = mlp_run<Ds...>(mlp, wa, ba, act, lane);
    int col = lane & 15, rbase = (lane >> 4) << 3;
#pragma unroll
    for (int r = 0; r < 8; ++r) {
        long rr = base + rbase + r;
        if (rr < nRows && col < OUTDIM) {
            float v = c[r]; v = v > 0.f ? v : 0.f;       // encoder ReLU
            out[rr * OUTSTRIDE + col] = v;
        }
    }
}

// Relational MLP over edges: m=[h[dst],h[src],e] -> e_new; atomic agg to dst.
template <int HD, int ED, int EDOUT, int... Ds>
__global__ __launch_bounds__(256)
void rel_kernel(MlpOff mlp, const __bf16* wa, const float* ba,
                const float* __restrict__ h, const float* __restrict__ eAll,
                int eInOff, int eOutOff, const int* __restrict__ ei,
                const float* __restrict__ emask, float* __restrict__ eOut,
                float* __restrict__ agg, int nE) {
    __shared__ alignas(32) __bf16 actS[LDSACT];
    int lane = threadIdx.x & 31, wave = threadIdx.x >> 5;
    __bf16* act = actS + wave * 16 * ACT_K;
    long base = ((long)blockIdx.x * WAVES + wave) * 16;
    const int* dst = ei + nE;
    int row = lane & 15;
    long e = base + row; if (e >= nE) e = nE - 1;
    int di = dst[e], si = ei[e];
    int j0 = (lane >> 4) << 4;
#pragma unroll
    for (int j = 0; j < 16; ++j) {
        int k = j0 + j;
        float v = 0.f;
        if (k < HD)               v = h[(long)di * HD + k];
        else if (k < 2 * HD)      v = h[(long)si * HD + (k - HD)];
        else if (k < 2 * HD + ED) v = eAll[e * 16 + eInOff + (k - 2 * HD)];
        act[row * ACT_K + k] = (__bf16)v;
    }
    v8f c = mlp_run<Ds...>(mlp, wa, ba, act, lane);
    int col = lane & 15, rbase = (lane >> 4) << 3;
    if (col < EDOUT) {
#pragma unroll
        for (int r = 0; r < 8; ++r) {
            long ee = base + rbase + r;
            if (ee < nE) {
                float v = c[r];
                if (eOut) eOut[ee * 16 + eOutOff + col] = v;
                float msg = emask ? v * emask[ee] : v;
                atomicAdd(&agg[(long)dst[ee] * EDOUT + col], msg);  // global_atomic_add_f32
            }
        }
    }
}

// Object MLP over nodes: in=[h,agg]; MODE0: h = a*h+(1-a)*relu(dh); MODE1: out=dh
template <int HD, int AGGC, int MODE, int... Ds>
__global__ __launch_bounds__(256)
void obj_kernel(MlpOff mlp, const __bf16* wa, const float* ba,
                float* __restrict__ h, const float* __restrict__ agg, int nN,
                float* __restrict__ outBuf, float alpha) {
    __shared__ alignas(32) __bf16 actS[LDSACT];
    int lane = threadIdx.x & 31, wave = threadIdx.x >> 5;
    __bf16* act = actS + wave * 16 * ACT_K;
    long base = ((long)blockIdx.x * WAVES + wave) * 16;
    int row = lane & 15;
    long nI = base + row; if (nI >= nN) nI = nN - 1;
    int j0 = (lane >> 4) << 4;
#pragma unroll
    for (int j = 0; j < 16; ++j) {
        int k = j0 + j;
        float v = 0.f;
        if (k < HD)             v = h[nI * HD + k];
        else if (k < HD + AGGC) v = agg[nI * AGGC + (k - HD)];
        act[row * ACT_K + k] = (__bf16)v;
    }
    v8f c = mlp_run<Ds...>(mlp, wa, ba, act, lane);
    int col = lane & 15, rbase = (lane >> 4) << 3;
#pragma unroll
    for (int r = 0; r < 8; ++r) {
        long nn = base + rbase + r;
        if (nn < nN) {
            if (MODE == 0 && col < HD) {
                float dh = c[r] > 0.f ? c[r] : 0.f;
                h[nn * HD + col] = alpha * h[nn * HD + col] + (1.f - alpha) * dh;
            } else if (MODE == 1 && col == 0) {
                outBuf[nn] = c[r];                      // track_params (raw)
            }
        }
    }
}

// W head: sigmoid(mlp(e_list)) -> edge_weights + 0/1 mask
__global__ __launch_bounds__(256)
void whead_kernel(MlpOff mlp, const __bf16* wa, const float* ba,
                  const float* __restrict__ eAll, int nE,
                  float* __restrict__ ew, float* __restrict__ emask) {
    __shared__ alignas(32) __bf16 actS[LDSACT];
    int lane = threadIdx.x & 31, wave = threadIdx.x >> 5;
    __bf16* act = actS + wave * 16 * ACT_K;
    long base = ((long)blockIdx.x * WAVES + wave) * 16;
    int row = lane & 15;
    long e = base + row; if (e >= nE) e = nE - 1;
    int j0 = (lane >> 4) << 4;
#pragma unroll
    for (int j = 0; j < 16; ++j) {
        int k = j0 + j;
        float v = (k < 16) ? eAll[e * 16 + k] : 0.f;
        act[row * ACT_K + k] = (__bf16)v;
    }
    v8f c = mlp_run<16, 40, 1>(mlp, wa, ba, act, lane);
    int col = lane & 15, rbase = (lane >> 4) << 3;
    if (col == 0) {
#pragma unroll
        for (int r = 0; r < 8; ++r) {
            long ee = base + rbase + r;
            if (ee < nE) {
                float s = 1.f / (1.f + __expf(-c[r]));
                ew[ee] = s;
                emask[ee] = (s > 0.5f) ? 1.f : 0.f;
            }
        }
    }
}

// B and X heads fused: beta = sigmoid(mlpB(h))+1e-5 ; hOut = mlpX(h)
__global__ __launch_bounds__(256)
void bx_kernel(MlpOff mB, MlpOff mX, const __bf16* wa, const float* ba,
               const float* __restrict__ h, int nN,
               float* __restrict__ beta, float* __restrict__ hOut) {
    __shared__ alignas(32) __bf16 actS[LDSACT];
    int lane = threadIdx.x & 31, wave = threadIdx.x >> 5;
    __bf16* act = actS + wave * 16 * ACT_K;
    long base = ((long)blockIdx.x * WAVES + wave) * 16;
    int row = lane & 15;
    long nI = base + row; if (nI >= nN) nI = nN - 1;
    int j0 = (lane >> 4) << 4;
#pragma unroll
    for (int j = 0; j < 16; ++j) {
        int k = j0 + j;
        act[row * ACT_K + k] = (__bf16)((k < 5) ? h[nI * 5 + k] : 0.f);
    }
    v8f cB = mlp_run<5, 40, 1>(mB, wa, ba, act, lane);
    int col = lane & 15, rbase = (lane >> 4) << 3;
    if (col == 0) {
#pragma unroll
        for (int r = 0; r < 8; ++r) {
            long nn = base + rbase + r;
            if (nn < nN) beta[nn] = 1.f / (1.f + __expf(-cB[r])) + 1e-5f;
        }
    }
    __syncthreads();
#pragma unroll
    for (int j = 0; j < 16; ++j) {                 // restage h for X head
        int k = j0 + j;
        act[row * ACT_K + k] = (__bf16)((k < 5) ? h[nI * 5 + k] : 0.f);
    }
    v8f cX = mlp_run<5, 40, 2>(mX, wa, ba, act, lane);
    if (col < 2) {
#pragma unroll
        for (int r = 0; r < 8; ++r) {
            long nn = base + rbase + r;
            if (nn < nN) hOut[nn * 2 + col] = cX[r];
        }
    }
}

// ---------------------------------------------------------------------------

extern "C" void kernel_launch(void* const* d_in, const int* in_sizes, int n_in,
                              void* d_out, int out_size, void* d_ws, size_t ws_size,
                              hipStream_t stream) {
    const float* x  = (const float*)d_in[0];
    const int*   ei = (const int*)d_in[1];
    const float* ea = (const float*)d_in[2];
    int nN = in_sizes[0] / 6;
    int nE = in_sizes[2] / 4;

    // ---- build linear-layer descriptors (JAX pytree leaf order) ----
    PackArgs pa;
    pa.n = 0;
    int wcur = 0, bcur = 0;
    auto addMlp = [&](int base, const int* dims, int nd) -> MlpOff {
        MlpOff m;
        for (int i = 0; i < nd - 1; ++i) {
            int K = dims[i], N = dims[i + 1];
            int Kp = (K + 31) & ~31, Np = (N + 15) & ~15;
            PackLin& L = pa.l[pa.n++];
            L.W = (const float*)d_in[base + 2 * i];
            L.B = (const float*)d_in[base + 2 * i + 1];
            L.K = K; L.N = N; L.woff = wcur; L.boff = bcur;
            m.woff[i] = wcur; m.boff[i] = bcur;
            wcur += Kp * Np; bcur += Np;
        }
        return m;
    };
    const int dEncN[3] = {6, 40, 5},  dEncE[3] = {4, 40, 4};
    const int dRel[5]  = {14, 40, 40, 40, 4}, dObj[5] = {9, 40, 40, 40, 5};
    const int dW[3]    = {16, 40, 1}, dB[3] = {5, 40, 1}, dX[3] = {5, 40, 2};
    const int dPrel[5] = {26, 40, 40, 40, 1}, dPobj[5] = {6, 40, 40, 40, 1};

    // params dict keys sorted: B, P{obj,rel}, W, X, ec_edge_enc,
    // ec_layers[l]{obj,rel}, ec_node_enc, hc_edge_enc, hc_layers, hc_node_enc
    MlpOff mB_   = addMlp(3,  dB, 3);
    MlpOff mPobj = addMlp(7,  dPobj, 5);
    MlpOff mPrel = addMlp(15, dPrel, 5);
    MlpOff mW_   = addMlp(23, dW, 3);
    MlpOff mX_   = addMlp(27, dX, 3);
    MlpOff mEncE_ec = addMlp(31, dEncE, 3);
    MlpOff mObjEC[3], mRelEC[3], mObjHC[3], mRelHC[3];
    for (int l = 0; l < 3; ++l) {
        mObjEC[l] = addMlp(35 + 16 * l, dObj, 5);
        mRelEC[l] = addMlp(43 + 16 * l, dRel, 5);
    }
    MlpOff mEncN_ec = addMlp(83, dEncN, 3);
    MlpOff mEncE_hc = addMlp(87, dEncE, 3);
    for (int l = 0; l < 3; ++l) {
        mObjHC[l] = addMlp(91 + 16 * l, dObj, 5);
        mRelHC[l] = addMlp(99 + 16 * l, dRel, 5);
    }
    MlpOff mEncN_hc = addMlp(139, dEncN, 3);

    // ---- carve workspace ----
    size_t off = 0;
    auto carve = [&](size_t bytes) -> void* {
        void* p = (char*)d_ws + off;
        off = (off + bytes + 255) & ~(size_t)255;
        return p;
    };
    __bf16* wArena = (__bf16*)carve((size_t)wcur * sizeof(__bf16));
    float*  bArena = (float*) carve((size_t)bcur * sizeof(float));
    float*  hEC    = (float*) carve((size_t)nN * 5 * sizeof(float));
    float*  hHC    = (float*) carve((size_t)nN * 5 * sizeof(float));
    float*  agg    = (float*) carve((size_t)nN * 4 * sizeof(float));
    float*  aggP   = (float*) carve((size_t)nN * sizeof(float));
    float*  eEC    = (float*) carve((size_t)nE * 16 * sizeof(float));
    float*  eHC    = (float*) carve((size_t)nE * 16 * sizeof(float));
    float*  emask  = (float*) carve((size_t)nE * sizeof(float));
    (void)ws_size; (void)n_in; (void)out_size;

    float* outF    = (float*)d_out;
    float* outEW   = outF;
    float* outH    = outF + (size_t)nE;
    float* outBeta = outF + (size_t)nE + 2 * (size_t)nN;
    float* outTrk  = outF + (size_t)nE + 3 * (size_t)nN;

    int nodeBlocks = (((nN + 15) / 16) + WAVES - 1) / WAVES;
    int edgeBlocks = (((nE + 15) / 16) + WAVES - 1) / WAVES;

    pack_kernel<<<pa.n, 256, 0, stream>>>(pa, wArena, bArena);

    // ---- edge-classifier branch ----
    enc_kernel<5, 5, 6, 40, 5><<<nodeBlocks, 256, 0, stream>>>(
        mEncN_ec, wArena, bArena, x, nN, hEC);
    enc_kernel<16, 4, 4, 40, 4><<<edgeBlocks, 256, 0, stream>>>(
        mEncE_ec, wArena, bArena, ea, nE, eEC);
    for (int l = 0; l < 3; ++l) {
        hipMemsetAsync(agg, 0, (size_t)nN * 4 * sizeof(float), stream);
        rel_kernel<5, 4, 4, 14, 40, 40, 40, 4><<<edgeBlocks, 256, 0, stream>>>(
            mRelEC[l], wArena, bArena, hEC, eEC, l * 4, (l + 1) * 4,
            ei, nullptr, eEC, agg, nE);
        obj_kernel<5, 4, 0, 9, 40, 40, 40, 5><<<nodeBlocks, 256, 0, stream>>>(
            mObjEC[l], wArena, bArena, hEC, agg, nN, nullptr, 0.5f);
    }
    whead_kernel<<<edgeBlocks, 256, 0, stream>>>(mW_, wArena, bArena, eEC, nE, outEW, emask);

    // ---- condenser branch (masked messages) ----
    enc_kernel<5, 5, 6, 40, 5><<<nodeBlocks, 256, 0, stream>>>(
        mEncN_hc, wArena, bArena, x, nN, hHC);
    enc_kernel<16, 4, 4, 40, 4><<<edgeBlocks, 256, 0, stream>>>(
        mEncE_hc, wArena, bArena, ea, nE, eHC);
    for (int l = 0; l < 3; ++l) {
        hipMemsetAsync(agg, 0, (size_t)nN * 4 * sizeof(float), stream);
        rel_kernel<5, 4, 4, 14, 40, 40, 40, 4><<<edgeBlocks, 256, 0, stream>>>(
            mRelHC[l], wArena, bArena, hHC, eHC, l * 4, (l + 1) * 4,
            ei, emask, eHC, agg, nE);
        obj_kernel<5, 4, 0, 9, 40, 40, 40, 5><<<nodeBlocks, 256, 0, stream>>>(
            mObjHC[l], wArena, bArena, hHC, agg, nN, nullptr, 0.5f);
    }

    // ---- heads ----
    bx_kernel<<<nodeBlocks, 256, 0, stream>>>(mB_, mX_, wArena, bArena, hHC, nN, outBeta, outH);
    hipMemsetAsync(aggP, 0, (size_t)nN * sizeof(float), stream);
    rel_kernel<5, 16, 1, 26, 40, 40, 40, 1><<<edgeBlocks, 256, 0, stream>>>(
        mPrel, wArena, bArena, hHC, eHC, 0, 0, ei, emask, nullptr, aggP, nE);
    obj_kernel<5, 1, 1, 6, 40, 40, 40, 1><<<nodeBlocks, 256, 0, stream>>>(
        mPobj, wArena, bArena, hHC, aggP, nN, outTrk, 0.5f);
}
